// GCN_4_86354612453996
// MI455X (gfx1250) — matrix operand order repaired and checked by reading, not compiled, not run on hardware
//
#include <hip/hip_runtime.h>
#include <hip/hip_bf16.h>

typedef __attribute__((ext_vector_type(16))) __bf16 v16bf;
typedef __attribute__((ext_vector_type(8)))  __bf16 v8bf;
typedef __attribute__((ext_vector_type(8)))  float  v8f;

// ---------------------------------------------------------------------------
// f32 -> bf16 elementwise convert (for GEMM A input)
// ---------------------------------------------------------------------------
__global__ void __launch_bounds__(256) k_f32_to_bf16(const float* __restrict__ in,
                                                     __bf16* __restrict__ out,
                                                     size_t n) {
    size_t i = (size_t)blockIdx.x * blockDim.x + threadIdx.x;
    if (i < n) out[i] = (__bf16)in[i];
}

// ---------------------------------------------------------------------------
// Pack W [Kin x Nout] (f32, row-major) into Wt [Npad x Kpad] bf16,
// column-major w.r.t. original (Wt[n][k] = W[k][n]), zero padded.
// This makes each lane's B-fragment one contiguous 32-byte chunk.
// ---------------------------------------------------------------------------
__global__ void __launch_bounds__(256) k_pack_wt(const float* __restrict__ W,
                                                 __bf16* __restrict__ Wt,
                                                 int Kin, int Nout,
                                                 int Kpad, int Npad) {
    int idx = blockIdx.x * blockDim.x + threadIdx.x;
    if (idx >= Kpad * Npad) return;
    int k = idx % Kpad;
    int n = idx / Kpad;
    float v = (k < Kin && n < Nout) ? W[(size_t)k * Nout + n] : 0.0f;
    Wt[(size_t)n * Kpad + k] = (__bf16)v;
}

// ---------------------------------------------------------------------------
// zero f32 buffer
// ---------------------------------------------------------------------------
__global__ void __launch_bounds__(256) k_zero_f32(float* __restrict__ p, size_t n) {
    size_t i = (size_t)blockIdx.x * blockDim.x + threadIdx.x;
    if (i < n) p[i] = 0.0f;
}

// ---------------------------------------------------------------------------
// WMMA GEMM: C[M x Nreal] = A[M x K](bf16) * Wt^T  (Wt stored [Npad x K] bf16,
// Wt[n][k] = W[k][n]).  Wave computes 16 rows x 64 cols (4 WMMA tiles sharing
// one A fragment). Block = 256 thr = 8 waves -> 128 x 64 tile.
// K multiple of 32; Npad multiple of 64; store guarded by Nreal.
// ---------------------------------------------------------------------------
__global__ void __launch_bounds__(256) k_wmma_gemm(const __bf16* __restrict__ A,
                                                   const __bf16* __restrict__ Wt,
                                                   float* __restrict__ C,
                                                   int M, int K, int Nreal) {
    const int lane = threadIdx.x & 31;
    const int wave = threadIdx.x >> 5;
    const int row0 = (blockIdx.x * 8 + wave) << 4;
    if (row0 >= M) return;                      // wave-uniform exit
    const int col0 = blockIdx.y << 6;

    const int r  = lane & 15;
    const int ka = (lane >> 4) << 4;            // A frag K base: 0 or 16
    const int kb = (lane >> 4) << 4;            // B frag K base: 0 or 16

    const __bf16* ap  = A  + (size_t)(row0 + r) * K + ka;
    const __bf16* bp0 = Wt + (size_t)(col0 +  0 + r) * K + kb;
    const __bf16* bp1 = Wt + (size_t)(col0 + 16 + r) * K + kb;
    const __bf16* bp2 = Wt + (size_t)(col0 + 32 + r) * K + kb;
    const __bf16* bp3 = Wt + (size_t)(col0 + 48 + r) * K + kb;

    v8f acc[4] = {v8f{}, v8f{}, v8f{}, v8f{}};

    for (int k = 0; k < K; k += 32) {
        // A fragment: row r, K = {ka..ka+15} contiguous
        v16bf a;
        {
            v8bf lo = *(const v8bf*)(ap + k);
            v8bf hi = *(const v8bf*)(ap + k + 8);
#pragma unroll
            for (int i = 0; i < 8; ++i) { a[i] = lo[i]; a[i + 8] = hi[i]; }
        }
        // B fragments: col r, K = {kb..kb+15} contiguous in packed Wt
        v16bf b0 = *(const v16bf*)(bp0 + k);
        v16bf b1 = *(const v16bf*)(bp1 + k);
        v16bf b2 = *(const v16bf*)(bp2 + k);
        v16bf b3 = *(const v16bf*)(bp3 + k);

        acc[0] = __builtin_amdgcn_wmma_f32_16x16x32_bf16(false, a, false, b0, (short)0, acc[0], false, false);
        acc[1] = __builtin_amdgcn_wmma_f32_16x16x32_bf16(false, a, false, b1, (short)0, acc[1], false, false);
        acc[2] = __builtin_amdgcn_wmma_f32_16x16x32_bf16(false, a, false, b2, (short)0, acc[2], false, false);
        acc[3] = __builtin_amdgcn_wmma_f32_16x16x32_bf16(false, a, false, b3, (short)0, acc[3], false, false);
    }

    // D layout: VGPR i -> row row0 + i + (lane<16?0:8), col col0 + t*16 + (lane&15)
    const int cn = lane & 15;
    const int rh = (lane >> 4) << 3;
#pragma unroll
    for (int t = 0; t < 4; ++t) {
        int c = col0 + t * 16 + cn;
        if (c < Nreal) {
#pragma unroll
            for (int i = 0; i < 8; ++i)
                C[(size_t)(row0 + rh + i) * Nreal + c] = acc[t][i];
        }
    }
}

// ---------------------------------------------------------------------------
// SpMM: agg[dst] += support[src] * ew, one wave per edge, lanes stride feats.
// Coalesced gather + coalesced no-return f32 atomics (global_atomic_add_f32).
// ---------------------------------------------------------------------------
__global__ void __launch_bounds__(256) k_spmm_edges(const float* __restrict__ sup,
                                                    const int* __restrict__ src,
                                                    const int* __restrict__ dst,
                                                    const float* __restrict__ ew,
                                                    float* __restrict__ agg,
                                                    int nE, int d) {
    int e    = (int)(((size_t)blockIdx.x * blockDim.x + threadIdx.x) >> 5);
    int lane = threadIdx.x & 31;
    if (e >= nE) return;
    int s = src[e];
    int t = dst[e];
    float w = ew[e];
    const float* srow = sup + (size_t)s * d;
    float*       trow = agg + (size_t)t * d;
    for (int f = lane; f < d; f += 32) {
        float v = srow[f] * w;
        (void)__hip_atomic_fetch_add(&trow[f], v, __ATOMIC_RELAXED, __HIP_MEMORY_SCOPE_AGENT);
    }
}

// ---------------------------------------------------------------------------
// h_next = bf16(relu(agg + b))   (feeds next layer's WMMA A operand)
// ---------------------------------------------------------------------------
__global__ void __launch_bounds__(256) k_bias_relu_bf16(const float* __restrict__ agg,
                                                        const float* __restrict__ b,
                                                        __bf16* __restrict__ out,
                                                        size_t total, int d) {
    size_t i = (size_t)blockIdx.x * blockDim.x + threadIdx.x;
    if (i >= total) return;
    int f = (int)(i % d);
    float v = agg[i] + b[f];
    out[i] = (__bf16)fmaxf(v, 0.0f);
}

// ---------------------------------------------------------------------------
// Final layer: out = log_softmax(relu(agg + b)), one wave per node (d == 40:
// lane covers class lane and lane+32). Wave32 shuffle reductions.
// ---------------------------------------------------------------------------
__global__ void __launch_bounds__(256) k_bias_relu_logsoftmax(const float* __restrict__ agg,
                                                              const float* __restrict__ b,
                                                              float* __restrict__ out,
                                                              int N, int d) {
    int n    = (int)(((size_t)blockIdx.x * blockDim.x + threadIdx.x) >> 5);
    int lane = threadIdx.x & 31;
    if (n >= N) return;
    const float* arow = agg + (size_t)n * d;
    float v0 = fmaxf(arow[lane] + b[lane], 0.0f);
    int  f1   = lane + 32;
    bool has1 = f1 < d;
    float v1 = has1 ? fmaxf(arow[f1] + b[f1], 0.0f) : -INFINITY;

    float m = fmaxf(v0, v1);
#pragma unroll
    for (int off = 16; off; off >>= 1) m = fmaxf(m, __shfl_xor(m, off, 32));
    float s = expf(v0 - m) + (has1 ? expf(v1 - m) : 0.0f);
#pragma unroll
    for (int off = 16; off; off >>= 1) s += __shfl_xor(s, off, 32);
    float ls = logf(s);

    float* orow = out + (size_t)n * d;
    orow[lane] = v0 - m - ls;
    if (has1) orow[f1] = v1 - m - ls;
}

// ---------------------------------------------------------------------------
// Orchestration
// ---------------------------------------------------------------------------
extern "C" void kernel_launch(void* const* d_in, const int* in_sizes, int n_in,
                              void* d_out, int out_size, void* d_ws, size_t ws_size,
                              hipStream_t stream) {
    const float* x    = (const float*)d_in[0];
    const int*   esrc = (const int*)  d_in[1];
    const int*   edst = (const int*)  d_in[2];
    const float* ew   = (const float*)d_in[3];
    const float* W[4]    = { (const float*)d_in[4], (const float*)d_in[6],
                             (const float*)d_in[8], (const float*)d_in[10] };
    const float* bias[4] = { (const float*)d_in[5], (const float*)d_in[7],
                             (const float*)d_in[9], (const float*)d_in[11] };
    float* out = (float*)d_out;

    const int NFEAT = 512;
    const int N  = in_sizes[0] / NFEAT;   // 100000
    const int nE = in_sizes[1];           // 3200000

    const int Kdim[4] = {512, 256, 128, 64};
    const int Nout[4] = {256, 128, 64, 40};
    const int Npad[4] = {256, 128, 64, 64};

    // workspace carve-up (256B aligned)
    char* ws = (char*)d_ws;
    size_t off = 0;
    auto take = [&](size_t bytes) -> void* {
        off = (off + 255) & ~(size_t)255;
        void* p = ws + off;
        off += bytes;
        return p;
    };
    __bf16* hb  = (__bf16*)take((size_t)N * 512 * sizeof(__bf16)); // GEMM A (bf16)
    float*  sup = (float*) take((size_t)N * 256 * sizeof(float));  // support
    float*  agg = (float*) take((size_t)N * 256 * sizeof(float));  // aggregated
    __bf16* Wt[4];
    for (int i = 0; i < 4; ++i)
        Wt[i] = (__bf16*)take((size_t)Npad[i] * Kdim[i] * sizeof(__bf16));

    // x -> bf16
    {
        size_t tot = (size_t)N * NFEAT;
        k_f32_to_bf16<<<(unsigned)((tot + 255) / 256), 256, 0, stream>>>(x, hb, tot);
    }

    for (int L = 0; L < 4; ++L) {
        const int K = Kdim[L], No = Nout[L], Np = Npad[L];

        // pack/transpose/pad W into bf16
        {
            int tot = Np * K;
            k_pack_wt<<<(tot + 255) / 256, 256, 0, stream>>>(W[L], Wt[L], K, No, K, Np);
        }
        // support = h @ W   (WMMA)
        {
            dim3 grid((N + 127) / 128, Np / 64);
            k_wmma_gemm<<<grid, 256, 0, stream>>>(hb, Wt[L], sup, N, K, No);
        }
        // agg = 0
        {
            size_t tot = (size_t)N * No;
            k_zero_f32<<<(unsigned)((tot + 255) / 256), 256, 0, stream>>>(agg, tot);
        }
        // agg += scatter(support[src] * ew)
        {
            k_spmm_edges<<<(nE + 7) / 8, 256, 0, stream>>>(sup, esrc, edst, ew, agg, nE, No);
        }
        // epilogue
        if (L < 3) {
            size_t tot = (size_t)N * No;
            k_bias_relu_bf16<<<(unsigned)((tot + 255) / 256), 256, 0, stream>>>(agg, bias[L], hb, tot, No);
        } else {
            k_bias_relu_logsoftmax<<<(N + 7) / 8, 256, 0, stream>>>(agg, bias[L], out, N, No);
        }
    }
}
